// SparseFF_51041391346160
// MI455X (gfx1250) — matrix operand order repaired and checked by reading, not compiled, not run on hardware
//
#include <hip/hip_runtime.h>

typedef __attribute__((ext_vector_type(16))) __bf16 v16bf;
typedef __attribute__((ext_vector_type(8)))  float  v8f;
typedef __attribute__((ext_vector_type(4)))  unsigned int ui32x4;
typedef __attribute__((ext_vector_type(8)))  int i32x8;
typedef __attribute__((ext_vector_type(4)))  int i32x4;

constexpr int kB  = 4;
constexpr int kT  = 2048;
constexpr int kBT = kB * kT;      // 8192 tokens
constexpr int kD  = 1024;
constexpr int kH  = 4096;
constexpr int kR  = 64;
constexpr int kS  = 4;
constexpr int kCH = kH / kS;      // 1024 per chunk

// ---------------------------------------------------------------------------
// bf16 helpers
// ---------------------------------------------------------------------------
__device__ __forceinline__ unsigned short f2bfbits(float f) {
  unsigned u = __builtin_bit_cast(unsigned, f);
  return (unsigned short)((u + 0x7FFFu + ((u >> 16) & 1u)) >> 16);
}

union Frag32 {                      // 32-byte fragment overlay
  uint4 q[2];
  v16bf f;
};

// A fragment (16x32 bf16) from row-major bf16 memory. Per lane: two
// contiguous 8-element (16 B) runs, K in [half*8, +8) and +16.
__device__ __forceinline__ v16bf load_a_frag_bf(const unsigned short* __restrict__ base,
                                                int stride, int row0, int k0) {
  int lane = threadIdx.x & 31;
  int m = lane & 15, half = lane >> 4;
  const unsigned short* rp = base + (size_t)(row0 + m) * stride + k0 + half * 8;
  Frag32 fr;
  fr.q[0] = *(const uint4*)(rp);
  fr.q[1] = *(const uint4*)(rp + 16);
  return fr.f;
}

// B fragment (32x16 bf16) for out = A*Wt (B[k][n] = W[n][k], W row-major bf16).
// Per lane: two contiguous 8-element (16 B) runs, K in [k0+half*8, +8) and +16.
__device__ __forceinline__ v16bf load_b_fragT_bf(const unsigned short* __restrict__ base,
                                                 int stride, int n0, int k0) {
  int lane = threadIdx.x & 31;
  int n = lane & 15, half = lane >> 4;
  const unsigned short* rp = base + (size_t)(n0 + n) * stride + k0 + half * 8;
  Frag32 fr;
  fr.q[0] = *(const uint4*)(rp);
  fr.q[1] = *(const uint4*)(rp + 16);
  return fr.f;
}

// ---------------------------------------------------------------------------
// TDM: async 2D tile load (16 rows x kCH floats, row stride kH floats) into
// LDS at byte offset lds_off. D# per CDNA5 ISA 8.3/8.4 (2D, groups 2/3 zero).
// This toolchain's builtin takes the 6-arg form (clang-23 / therock headers).
// ---------------------------------------------------------------------------
__device__ __forceinline__ void tdm_load_gumbel_tile(const float* gsrc, unsigned lds_off) {
  unsigned long long ga = (unsigned long long)(size_t)gsrc;
  ui32x4 g0 = {
    1u,                                                // count=1, no gather
    lds_off,                                           // lds_addr (bytes)
    (unsigned)(ga & 0xFFFFFFFFu),                      // global_addr[31:0]
    (unsigned)((ga >> 32) & 0x1FFFFFFu) | (2u << 30)   // addr[56:32] | type=2
  };
  i32x8 g1 = {
    (int)(2u << 16),                 // workgroup_mask=0, data_size=2 (4 bytes)
    (int)((unsigned)kCH << 16),      // tensor_dim0[15:0] (bits 63:48) = 1024
    (int)(16u << 16),                // tensor_dim0 hi=0 | tensor_dim1 = 16
    (int)((unsigned)kCH << 16),      // tensor_dim1 hi=0 | tile_dim0 = 1024
    16,                              // tile_dim1 = 16, tile_dim2 = 0
    kH,                              // tensor_dim0_stride = 4096 (elements)
    0,                               // stride hi | tensor_dim1_stride lo (unused)
    0
  };
  i32x4 zz4 = {0, 0, 0, 0};
  i32x8 zz8 = {0, 0, 0, 0, 0, 0, 0, 0};
  __builtin_amdgcn_tensor_load_to_lds(g0, g1, zz4, zz4, zz8, 0);
}

// ---------------------------------------------------------------------------
// Kernel C: f32 -> packed bf16 (2 per dword), grid-stride, coalesced
// ---------------------------------------------------------------------------
__global__ __launch_bounds__(256) void k_cvt_bf16(const float* __restrict__ src,
                                                  unsigned int* __restrict__ dst,
                                                  int n2) {
  int i = blockIdx.x * 256 + threadIdx.x;
  int stride = gridDim.x * 256;
  for (; i < n2; i += stride) {
    float2 f = ((const float2*)src)[i];
    unsigned lo = f2bfbits(f.x);
    unsigned hi = f2bfbits(f.y);
    dst[i] = lo | (hi << 16);
  }
}

// ---------------------------------------------------------------------------
// Kernel 0: transpose W2 (D x H) -> W2t (H x D), fp32 (value-critical path)
// ---------------------------------------------------------------------------
__global__ __launch_bounds__(256) void k_transpose_w2(const float* __restrict__ W2,
                                                      float* __restrict__ W2t) {
  __shared__ float tile[32][33];
  int bx = blockIdx.x;            // along H
  int by = blockIdx.y;            // along D
  int tx = threadIdx.x & 31;
  int ty = threadIdx.x >> 5;      // 0..7
#pragma unroll
  for (int j = 0; j < 4; ++j) {
    int d = by * 32 + ty + j * 8;
    tile[ty + j * 8][tx] = W2[(size_t)d * kH + bx * 32 + tx];
  }
  __syncthreads();
#pragma unroll
  for (int j = 0; j < 4; ++j) {
    int h = bx * 32 + ty + j * 8;
    W2t[(size_t)h * kD + by * 32 + tx] = tile[tx][ty + j * 8];
  }
}

// ---------------------------------------------------------------------------
// Kernel 1: ch = relu(x @ Wc1ᵀ + bc1)  [BT x 64], bf16 WMMA, K = 1024.
// 128 threads = 4 waves; wave w owns 16 rows x all of N=64. All fragment
// loads of a K-step are issued before the 4 WMMAs so they can overlap.
// ---------------------------------------------------------------------------
__global__ __launch_bounds__(128) void k_controller1(const unsigned short* __restrict__ xb,
                                                     const unsigned short* __restrict__ wc1b,
                                                     const float* __restrict__ bc1,
                                                     unsigned short* __restrict__ chb) {
  int wave = threadIdx.x >> 5;
  int lane = threadIdx.x & 31;
  int row0 = blockIdx.x * 64 + wave * 16;

  v8f acc[4] = {{}, {}, {}, {}};
  for (int k0 = 0; k0 < kD; k0 += 32) {
    v16bf a = load_a_frag_bf(xb, kD, row0, k0);
    v16bf b[4];
#pragma unroll
    for (int nt = 0; nt < 4; ++nt) b[nt] = load_b_fragT_bf(wc1b, kD, nt * 16, k0);
#pragma unroll
    for (int nt = 0; nt < 4; ++nt) {
      acc[nt] = __builtin_amdgcn_wmma_f32_16x16x32_bf16(false, a, false, b[nt],
                                                        (short)0, acc[nt], false, false);
    }
  }
  int half = lane >> 4, nc = lane & 15;
#pragma unroll
  for (int nt = 0; nt < 4; ++nt) {
    int n = nt * 16 + nc;
    float bias = bc1[n];
#pragma unroll
    for (int v = 0; v < 8; ++v) {
      int m = v + 8 * half;
      float t = acc[nt][v] + bias;
      t = t > 0.f ? t : 0.f;
      chb[(size_t)(row0 + m) * kR + n] = f2bfbits(t);
    }
  }
}

// ---------------------------------------------------------------------------
// Kernel 2: logits = ch @ Wc2ᵀ (bf16 WMMA) -> LDS z-buffer. Gumbel tiles are
// double-buffered into LDS by the Tensor Data Mover, overlapped with the
// WMMAs of the current chunk. Reduction: pass 1 adds bc2 + gumbel (LDS) and
// tracks argmax; pass 2 computes the exp-sum -> mask value (1-p)+p.
// Dynamic LDS: [0,64K) z | [64K,128K) gbuf0 | [128K,192K) gbuf1.
// ---------------------------------------------------------------------------
__global__ __launch_bounds__(256) void k_router(const unsigned short* __restrict__ chb,
                                                const unsigned short* __restrict__ wc2b,
                                                const float* __restrict__ bc2,
                                                const float* __restrict__ gumbel,
                                                int* __restrict__ idx_out,
                                                float* __restrict__ mval_out) {
  extern __shared__ float lds[];          // z: lds[0..16383], gbuf: +16384, +32768
  constexpr unsigned kGOffB = 16 * kCH * 4;   // 65536 bytes
  int mt   = blockIdx.x;
  int wave = threadIdx.x >> 5;
  int lane = threadIdx.x & 31;
  int half = lane >> 4;
  int nc   = lane & 15;

  const float* gbase = gumbel + (size_t)(mt * 16) * kH;

  // kick off the TDM load of chunk 0's gumbel tile into gbuf0
  if (wave == 0) {
    tdm_load_gumbel_tile(gbase, kGOffB);
  }

  // A fragments (16 tokens x K=64) stay resident across all 256 N-tiles
  v16bf a0 = load_a_frag_bf(chb, kR, mt * 16, 0);
  v16bf a1 = load_a_frag_bf(chb, kR, mt * 16, 32);

  for (int s = 0; s < kS; ++s) {
    for (int nt = wave; nt < kCH / 16; nt += 8) {
      int h0 = s * kCH + nt * 16;
      v16bf b0 = load_b_fragT_bf(wc2b, kR, h0, 0);
      v16bf b1 = load_b_fragT_bf(wc2b, kR, h0, 32);
      v8f acc = {};
      acc = __builtin_amdgcn_wmma_f32_16x16x32_bf16(false, a0, false, b0,
                                                    (short)0, acc, false, false);
      acc = __builtin_amdgcn_wmma_f32_16x16x32_bf16(false, a1, false, b1,
                                                    (short)0, acc, false, false);
#pragma unroll
      for (int v = 0; v < 8; ++v) {
        int m = v + 8 * half;
        lds[m * kCH + nt * 16 + nc] = acc[v];
      }
    }
    // prefetch next chunk's gumbel tile; make sure this chunk's tile landed
    if (wave == 0) {
      if (s + 1 < kS) {
        tdm_load_gumbel_tile(gbase + (s + 1) * kCH,
                             kGOffB + (unsigned)(((s + 1) & 1) * kGOffB));
        __builtin_amdgcn_s_wait_tensorcnt(1);
      } else {
        __builtin_amdgcn_s_wait_tensorcnt(0);
      }
    }
    __syncthreads();
    // 8 waves x 2 rows each
    const float* gtile = lds + 16 * kCH * (1 + (s & 1));
    for (int r = wave * 2; r < wave * 2 + 2; ++r) {
      float* zr = lds + r * kCH;
      const float* grow = gtile + r * kCH;
      const float* brow = bc2 + s * kCH;
      int tok = mt * 16 + r;
      float best = -__builtin_inff();
      int bidx = 0;
      for (int c = lane; c < kCH; c += 32) {
        float v = zr[c] + brow[c] + grow[c];
        zr[c] = v;
        if (v > best) { best = v; bidx = c; }
      }
#pragma unroll
      for (int off = 16; off > 0; off >>= 1) {
        float ov = __shfl_xor(best, off, 32);
        int   oi = __shfl_xor(bidx, off, 32);
        if (ov > best || (ov == best && oi < bidx)) { best = ov; bidx = oi; }
      }
      float esum = 0.f;
      for (int c = lane; c < kCH; c += 32) esum += __expf(zr[c] - best);
#pragma unroll
      for (int off = 16; off > 0; off >>= 1) esum += __shfl_xor(esum, off, 32);
      if (lane == 0) {
        float p = 1.f / esum;             // softmax prob at the argmax entry
        idx_out[tok * kS + s]  = bidx;
        mval_out[tok * kS + s] = (1.0f - p) + p;   // straight-through value
      }
    }
    __syncthreads();
  }
}

// ---------------------------------------------------------------------------
// Kernel 3: sparse FFN. One block per token: 4 length-1024 fp32 dots against
// the selected W1 rows, then out = b2 + sum_s h_s * W2t[u_s,:].
// W2t rows are prefetched (global_prefetch_b8) while the dots run.
// ---------------------------------------------------------------------------
__global__ __launch_bounds__(256) void k_sparse_ffn(const float* __restrict__ x,
                                                    const float* __restrict__ W1,
                                                    const float* __restrict__ b1,
                                                    const float* __restrict__ W2t,
                                                    const float* __restrict__ b2,
                                                    const int* __restrict__ idx,
                                                    const float* __restrict__ mval,
                                                    float* __restrict__ out) {
  __shared__ float xs[kD];
  __shared__ float wred[8][kS];
  __shared__ float hv[kS];
  int tok  = blockIdx.x;
  int tid  = threadIdx.x;
  int wave = tid >> 5;
  int lane = tid & 31;

  for (int i = tid; i < kD; i += 256) xs[i] = x[(size_t)tok * kD + i];
  int u[kS];
#pragma unroll
  for (int s = 0; s < kS; ++s) u[s] = s * kCH + idx[tok * kS + s];

  // warm the 4 selected W2t rows (4 x 4 KB) while we compute the dots
#pragma unroll
  for (int s = 0; s < kS; ++s) {
    __builtin_prefetch(W2t + (size_t)u[s] * kD + tid * 4, 0, 0);
  }
  __syncthreads();

  float p[kS] = {0.f, 0.f, 0.f, 0.f};
  for (int i = tid; i < kD; i += 256) {
    float xv = xs[i];
#pragma unroll
    for (int s = 0; s < kS; ++s) p[s] += xv * W1[(size_t)u[s] * kD + i];
  }
#pragma unroll
  for (int s = 0; s < kS; ++s) {
#pragma unroll
    for (int off = 16; off > 0; off >>= 1) p[s] += __shfl_xor(p[s], off, 32);
  }
  if (lane == 0) {
#pragma unroll
    for (int s = 0; s < kS; ++s) wred[wave][s] = p[s];
  }
  __syncthreads();
  if (tid < kS) {
    float d = 0.f;
#pragma unroll
    for (int w = 0; w < 8; ++w) d += wred[w][tid];
    d += b1[u[tid]];
    d = d > 0.f ? d : 0.f;
    hv[tid] = d * mval[tok * kS + tid];
  }
  __syncthreads();
  for (int dd = tid; dd < kD; dd += 256) {
    float acc = b2[dd];
#pragma unroll
    for (int s = 0; s < kS; ++s) acc += hv[s] * W2t[(size_t)u[s] * kD + dd];
    out[(size_t)tok * kD + dd] = acc;
  }
}

// ---------------------------------------------------------------------------
extern "C" void kernel_launch(void* const* d_in, const int* in_sizes, int n_in,
                              void* d_out, int out_size, void* d_ws, size_t ws_size,
                              hipStream_t stream) {
  (void)in_sizes; (void)n_in; (void)out_size; (void)ws_size;
  const float* x      = (const float*)d_in[0];
  const float* W1     = (const float*)d_in[1];
  const float* b1     = (const float*)d_in[2];
  const float* W2     = (const float*)d_in[3];
  const float* b2     = (const float*)d_in[4];
  const float* Wc1    = (const float*)d_in[5];
  const float* bc1    = (const float*)d_in[6];
  const float* Wc2    = (const float*)d_in[7];
  const float* bc2    = (const float*)d_in[8];
  const float* gumbel = (const float*)d_in[9];
  float* out = (float*)d_out;

  // workspace layout (bytes):
  //   W2t  f32  [H x D]   @ 0          16777216
  //   xb   bf16 [BT x D]  @ 16777216   16777216
  //   wc1b bf16 [R x D]   @ 33554432     131072
  //   wc2b bf16 [H x R]   @ 33685504     524288
  //   chb  bf16 [BT x R]  @ 34209792    1048576
  //   idx  i32  [BT x S]  @ 35258368     131072
  //   mval f32  [BT x S]  @ 35389440     131072
  char* ws = (char*)d_ws;
  float*          W2t  = (float*)ws;
  unsigned short* xb   = (unsigned short*)(ws + 16777216u);
  unsigned short* wc1b = (unsigned short*)(ws + 33554432u);
  unsigned short* wc2b = (unsigned short*)(ws + 33685504u);
  unsigned short* chb  = (unsigned short*)(ws + 34209792u);
  int*            idx  = (int*)(ws + 35258368u);
  float*          mval = (float*)(ws + 35389440u);

  // one-shot conversions / transpose
  k_cvt_bf16<<<4096, 256, 0, stream>>>(x,   (unsigned int*)xb,   kBT * kD / 2);
  k_cvt_bf16<<<128,  256, 0, stream>>>(Wc1, (unsigned int*)wc1b, kR * kD / 2);
  k_cvt_bf16<<<512,  256, 0, stream>>>(Wc2, (unsigned int*)wc2b, kH * kR / 2);
  k_transpose_w2<<<dim3(kH / 32, kD / 32), 256, 0, stream>>>(W2, W2t);

  // controller + router + sparse FFN
  k_controller1<<<kBT / 64, 128, 0, stream>>>(xb, wc1b, bc1, chb);
  size_t shmem = (size_t)3 * 16 * kCH * sizeof(float);   // z + 2 gumbel buffers
  k_router<<<kBT / 16, 256, shmem, stream>>>(chb, wc2b, bc2, gumbel, idx, mval);
  k_sparse_ffn<<<kBT, 256, 0, stream>>>(x, W1, b1, W2t, b2, idx, mval, out);
}
